// MOELayer_6236292514565
// MI455X (gfx1250) — matrix operand order repaired and checked by reading, not compiled, not run on hardware
//
#include <hip/hip_runtime.h>

// ---------------------------------------------------------------------------
// GShard top-2 MoE for MI455X (gfx1250, wave32, WMMA).
// FFN GEMMs (>99% of 275 GFLOP) run on v_wmma_f32_16x16x32_bf16 with fp32
// accumulation. Weights are converted to bf16 AND pre-transposed to [N][K]
// so both A and B tile staging are contiguous b128 chunks, moved with
// global_load_async_to_lds_b128 (ASYNCcnt) into double-buffered LDS tiles.
// ---------------------------------------------------------------------------

typedef __attribute__((ext_vector_type(16))) __bf16        v16bf;
typedef __attribute__((ext_vector_type(8)))  float         v8f;
typedef __attribute__((ext_vector_type(4)))  unsigned int  v4u;
typedef __attribute__((ext_vector_type(8)))  unsigned int  v8u;

namespace {
constexpr int kS   = 8192;   // tokens = B*S
constexpr int kM   = 1024;   // d_model
constexpr int kE   = 8;      // experts
constexpr int kH   = 4096;   // d_ff
constexpr int kCap = 2048;   // per-expert capacity
}

__device__ __forceinline__ unsigned short f2bf(float f) {
  unsigned int u = __float_as_uint(f);
  u += 0x7FFFu + ((u >> 16) & 1u);           // round-to-nearest-even
  return (unsigned short)(u >> 16);
}

// Async global -> LDS copy, 16 bytes per lane. Tracked by ASYNCcnt.
__device__ __forceinline__ void async_copy_b128(unsigned lds_byte_off,
                                                const unsigned short* g) {
  asm volatile("global_load_async_to_lds_b128 %0, %1, off"
               :: "v"(lds_byte_off),
                  "v"((unsigned long long)(size_t)g)
               : "memory");
}
__device__ __forceinline__ void wait_async0() {
#if __has_builtin(__builtin_amdgcn_s_wait_asynccnt)
  __builtin_amdgcn_s_wait_asynccnt(0);
#else
  asm volatile("s_wait_asynccnt 0x0" ::: "memory");
#endif
}

// ---------------------------------------------------------------------------
// init: zero the gating accumulators (ws is poisoned before timing).
// ---------------------------------------------------------------------------
__global__ void init_kernel(float* __restrict__ sumg, int* __restrict__ cnt1) {
  int t = threadIdx.x;
  if (t < kE) { sumg[t] = 0.f; cnt1[t] = 0; }
}

// ---------------------------------------------------------------------------
// Fused fp32->bf16 convert + transpose: in [K,N] fp32 -> out [N,K] bf16.
// LDS-tiled 64x64 so both global sides are coalesced.
// ---------------------------------------------------------------------------
__global__ __launch_bounds__(256)
void cvtT_bf16(const float* __restrict__ in, unsigned short* __restrict__ out,
               int K, int N) {
  constexpr int T = 64, STR = 72;            // padded LDS row stride
  __shared__ unsigned short tile[T * STR];
  const int e = blockIdx.z;
  const float* gin = in + (size_t)e * K * N;
  unsigned short* gout = out + (size_t)e * N * K;
  const int kBase = blockIdx.y * T;
  const int nBase = blockIdx.x * T;
  const int t = threadIdx.x;
#pragma unroll
  for (int i = 0; i < 16; ++i) {
    int idx = (i << 8) + t;                  // 0..4095
    int k = idx >> 6, n = idx & 63;
    tile[k * STR + n] = f2bf(gin[(size_t)(kBase + k) * N + (nBase + n)]);
  }
  __syncthreads();
#pragma unroll
  for (int i = 0; i < 16; ++i) {
    int idx = (i << 8) + t;
    int n = idx >> 6, k = idx & 63;
    gout[(size_t)(nBase + n) * K + (kBase + k)] = tile[k * STR + n];
  }
}

// ---------------------------------------------------------------------------
// Gating: one wave per token. logits = x[s]@wg, softmax(8), top1/top2.
// ---------------------------------------------------------------------------
__global__ __launch_bounds__(256)
void gate_kernel(const float* __restrict__ x, const float* __restrict__ wg,
                 int* __restrict__ idx1, int* __restrict__ idx2,
                 float* __restrict__ g1, float* __restrict__ g2,
                 float* __restrict__ sumg, int* __restrict__ cnt1) {
  int lane = threadIdx.x & 31;
  int tok  = blockIdx.x * 8 + (threadIdx.x >> 5);
  const float* xr = x + (size_t)tok * kM;

  float acc[kE];
#pragma unroll
  for (int e = 0; e < kE; ++e) acc[e] = 0.f;

  for (int m = lane; m < kM; m += 32) {
    float xv = xr[m];
    const float* wr = wg + (size_t)m * kE;
#pragma unroll
    for (int e = 0; e < kE; ++e) acc[e] += xv * wr[e];
  }
#pragma unroll
  for (int e = 0; e < kE; ++e)
#pragma unroll
    for (int off = 16; off > 0; off >>= 1)
      acc[e] += __shfl_xor(acc[e], off, 32);

  if (lane == 0) {
    float mx = acc[0];
#pragma unroll
    for (int e = 1; e < kE; ++e) mx = fmaxf(mx, acc[e]);
    float g[kE], sum = 0.f;
#pragma unroll
    for (int e = 0; e < kE; ++e) { g[e] = __expf(acc[e] - mx); sum += g[e]; }
    float inv = 1.f / sum;
#pragma unroll
    for (int e = 0; e < kE; ++e) g[e] *= inv;

    int i1 = 0;
#pragma unroll
    for (int e = 1; e < kE; ++e) if (g[e] > g[i1]) i1 = e;   // first-max tiebreak
    int i2 = (i1 == 0) ? 1 : 0;
#pragma unroll
    for (int e = 0; e < kE; ++e)
      if (e != i1 && acc[e] > acc[i2]) i2 = e;

    idx1[tok] = i1; idx2[tok] = i2;
    g1[tok] = g[i1]; g2[tok] = g[i2];
#pragma unroll
    for (int e = 0; e < kE; ++e) atomicAdd(&sumg[e], g[e]);
    atomicAdd(&cnt1[i1], 1);
  }
}

// ---------------------------------------------------------------------------
// Scan: order-dependent cumsum positions + capacity drop + renorm weights.
// ---------------------------------------------------------------------------
__global__ void scan_kernel(const int* __restrict__ idx1, const int* __restrict__ idx2,
                            const float* __restrict__ g1, const float* __restrict__ g2,
                            const float* __restrict__ sumg, const int* __restrict__ cnt1,
                            int* __restrict__ slot1, int* __restrict__ slot2,
                            float* __restrict__ cw1, float* __restrict__ cw2,
                            int* __restrict__ src, float* __restrict__ laux_out) {
  if (threadIdx.x != 0 || blockIdx.x != 0) return;
  for (int i = 0; i < kE * kCap; ++i) src[i] = -1;

  int c1[kE], c2[kE];
#pragma unroll
  for (int e = 0; e < kE; ++e) { c1[e] = 0; c2[e] = 0; }

  for (int s = 0; s < kS; ++s) {
    int e1 = idx1[s], e2 = idx2[s];
    int p1 = c1[e1]++;                    // loc1
    int p2 = (c2[e2]++) + cnt1[e2];       // loc2 = cumsum(mask2)-1 + sum(mask1)
    bool k1 = p1 < kCap, k2 = p2 < kCap;
    float a = k1 ? g1[s] : 0.f;
    float b = k2 ? g2[s] : 0.f;
    float d = fmaxf(a + b, 1e-9f);
    float w1 = a / d, w2 = b / d;
    if (k1) { src[e1 * kCap + p1] = s; slot1[s] = e1 * kCap + p1; cw1[s] = w1; }
    else    { slot1[s] = 0; cw1[s] = 0.f; }
    if (k2) { src[e2 * kCap + p2] = s; slot2[s] = e2 * kCap + p2; cw2[s] = w2; }
    else    { slot2[s] = 0; cw2[s] = 0.f; }
  }

  float l = 0.f;
#pragma unroll
  for (int e = 0; e < kE; ++e)
    l += (sumg[e] / (float)kS) * ((float)cnt1[e] / (float)kS);
  *laux_out = l * (float)kE;   // mean_e(me*ce)*E^2 = E*sum_e(me*ce)
}

// ---------------------------------------------------------------------------
// Gather tokens into expert buffers Xe[e,c,:] (bf16). Empty slots -> 0.
// ---------------------------------------------------------------------------
__global__ __launch_bounds__(256)
void gather_kernel(const float* __restrict__ x, const int* __restrict__ src,
                   unsigned short* __restrict__ xe) {
  int slot = blockIdx.x;                 // 0 .. E*Cap-1
  int s = src[slot];
  unsigned short* dst = xe + (size_t)slot * kM;
  if (s < 0) {
    for (int m = threadIdx.x; m < kM; m += 256) dst[m] = 0;
  } else {
    const float* xr = x + (size_t)s * kM;
    for (int m = threadIdx.x; m < kM; m += 256) dst[m] = f2bf(xr[m]);
  }
}

// ---------------------------------------------------------------------------
// Tiled bf16 GEMM on v_wmma_f32_16x16x32_bf16.
//   C[e] = act(A[e] (Mr x K) @ Bt[e]^T + bias[e]),  Bt stored [N x K] bf16.
// Block: 256 threads (8 waves), tile 128x128, wave tile 32x64 (2x4 WMMA accs).
// Both A and Bt tiles are [128 rows x 32 K] row-major -> identical staging:
// global_load_async_to_lds_b128 into double-buffered LDS, overlapped with
// the WMMAs of the current tile (ASYNCcnt pipeline).
// Fragment layouts (16-bit):
//   A 16x32: lane<16 -> K 0-7/16-23, lane>=16 -> K 8-15/24-31
//   B 32x16: lane<16 -> K 0-7/16-23, lane>=16 -> K 8-15/24-31
// ---------------------------------------------------------------------------
template <bool RELU_BF16_OUT>
__global__ __launch_bounds__(256)
void moe_gemm(const unsigned short* __restrict__ A,
              const unsigned short* __restrict__ Bt,
              const float* __restrict__ bias,
              void* __restrict__ C,
              int Mr, int K, int N) {
  constexpr int SA = 40;                           // padded row stride (elems)
  __shared__ unsigned short As[2][128 * SA];
  __shared__ unsigned short Bs[2][128 * SA];

  const int e = blockIdx.z;
  const unsigned short* Ag = A  + (size_t)e * Mr * K + (size_t)blockIdx.y * 128 * K;
  const unsigned short* Bg = Bt + (size_t)e * N  * K + (size_t)blockIdx.x * 128 * K;

  const int t    = threadIdx.x;
  const int lane = t & 31;
  const int wid  = t >> 5;
  const int wm   = wid & 3;                        // 4 row groups of 32
  const int wn   = wid >> 2;                       // 2 col groups of 64
  const int r16  = lane & 15;
  const int khA  = (lane >> 4) << 3;               // 0 or 8  (A layout)
  const int khB  = (lane >> 4) << 3;               // 0 or 8  (B layout)

  // Per-thread staging chunk coordinates (2 x b128 per matrix per tile).
  const int row0 = (t * 2) >> 2;                   // chunk row
  const int kc0  = ((t * 2) & 3) << 3;             // 0,8,16,24
  const int row1 = (t * 2 + 1) >> 2;
  const int kc1  = ((t * 2 + 1) & 3) << 3;

  v8f acc[2][4];
#pragma unroll
  for (int a = 0; a < 2; ++a)
#pragma unroll
    for (int b = 0; b < 4; ++b)
      acc[a][b] = v8f{0.f, 0.f, 0.f, 0.f, 0.f, 0.f, 0.f, 0.f};

  auto stage = [&](int k0, int buf) {
    async_copy_b128((unsigned)(size_t)&As[buf][row0 * SA + kc0],
                    Ag + (size_t)row0 * K + (k0 + kc0));
    async_copy_b128((unsigned)(size_t)&As[buf][row1 * SA + kc1],
                    Ag + (size_t)row1 * K + (k0 + kc1));
    async_copy_b128((unsigned)(size_t)&Bs[buf][row0 * SA + kc0],
                    Bg + (size_t)row0 * K + (k0 + kc0));
    async_copy_b128((unsigned)(size_t)&Bs[buf][row1 * SA + kc1],
                    Bg + (size_t)row1 * K + (k0 + kc1));
  };

  const int nsteps = K >> 5;
  stage(0, 0);
  wait_async0();
  __syncthreads();

  for (int kt = 0; kt < nsteps; ++kt) {
    const int buf = kt & 1;
    if (kt + 1 < nsteps) stage((kt + 1) << 5, buf ^ 1);   // prefetch next tile

    v16bf af[2], bfr[4];
#pragma unroll
    for (int a = 0; a < 2; ++a) {
      const unsigned short* p = &As[buf][(wm * 32 + a * 16 + r16) * SA + khA];
      v4u lo = *(const v4u*)p;
      v4u hi = *(const v4u*)(p + 16);
      af[a] = __builtin_bit_cast(
          v16bf, (v8u)__builtin_shufflevector(lo, hi, 0, 1, 2, 3, 4, 5, 6, 7));
    }
#pragma unroll
    for (int b = 0; b < 4; ++b) {
      const unsigned short* p = &Bs[buf][(wn * 64 + b * 16 + r16) * SA + khB];
      v4u lo = *(const v4u*)p;
      v4u hi = *(const v4u*)(p + 16);
      bfr[b] = __builtin_bit_cast(
          v16bf, (v8u)__builtin_shufflevector(lo, hi, 0, 1, 2, 3, 4, 5, 6, 7));
    }
#pragma unroll
    for (int a = 0; a < 2; ++a)
#pragma unroll
      for (int b = 0; b < 4; ++b)
        acc[a][b] = __builtin_amdgcn_wmma_f32_16x16x32_bf16(
            false, af[a], false, bfr[b], (short)0, acc[a][b], false, false);

    wait_async0();          // next tile landed in the other buffer
    __syncthreads();        // everyone done reading this buffer
  }

  // ---- epilogue: bias (+ReLU->bf16 | ->fp32) ----
  const size_t coff = (size_t)e * Mr * N;
  const int rowBase = blockIdx.y * 128 + wm * 32 + ((lane >> 4) << 3);
  const int colBase = blockIdx.x * 128 + wn * 64 + (lane & 15);
  const float* bsE = bias + (size_t)e * N;
#pragma unroll
  for (int a = 0; a < 2; ++a) {
#pragma unroll
    for (int b = 0; b < 4; ++b) {
      int col = colBase + b * 16;
      float bv = bsE[col];
#pragma unroll
      for (int r = 0; r < 8; ++r) {
        int row = rowBase + a * 16 + r;
        float v = acc[a][b][r] + bv;
        if (RELU_BF16_OUT) {
          v = fmaxf(v, 0.f);
          ((unsigned short*)C)[coff + (size_t)row * N + col] = f2bf(v);
        } else {
          ((float*)C)[coff + (size_t)row * N + col] = v;
        }
      }
    }
  }
}

// ---------------------------------------------------------------------------
// Combine: out[s] = w1*eo[slot1[s]] + w2*eo[slot2[s]]
// ---------------------------------------------------------------------------
__global__ __launch_bounds__(256)
void combine_kernel(const float* __restrict__ eo,
                    const int* __restrict__ slot1, const int* __restrict__ slot2,
                    const float* __restrict__ cw1, const float* __restrict__ cw2,
                    float* __restrict__ out) {
  int tok = blockIdx.x;
  int s1 = slot1[tok], s2 = slot2[tok];
  float w1 = cw1[tok], w2 = cw2[tok];
  const float* e1 = eo + (size_t)s1 * kM;
  const float* e2 = eo + (size_t)s2 * kM;
  float* o = out + (size_t)tok * kM;
  for (int m = threadIdx.x; m < kM; m += 256)
    o[m] = w1 * e1[m] + w2 * e2[m];
}

// ---------------------------------------------------------------------------
extern "C" void kernel_launch(void* const* d_in, const int* in_sizes, int n_in,
                              void* d_out, int out_size, void* d_ws, size_t ws_size,
                              hipStream_t stream) {
  (void)in_sizes; (void)n_in; (void)out_size; (void)ws_size;

  const float* x  = (const float*)d_in[0];   // [8192, 1024]
  const float* wg = (const float*)d_in[1];   // [1024, 8]
  const float* w1 = (const float*)d_in[2];   // [8, 1024, 4096]
  const float* b1 = (const float*)d_in[3];   // [8, 4096]
  const float* w2 = (const float*)d_in[4];   // [8, 4096, 1024]
  const float* b2 = (const float*)d_in[5];   // [8, 1024]
  float* out = (float*)d_out;                // [8192*1024] + l_aux

  char* ws = (char*)d_ws;
  size_t off = 0;
  auto carve = [&](size_t bytes) -> void* {
    void* p = ws + off;
    off = (off + bytes + 255) & ~(size_t)255;
    return p;
  };

  int*   idx1  = (int*)  carve((size_t)kS * 4);
  int*   idx2  = (int*)  carve((size_t)kS * 4);
  float* g1    = (float*)carve((size_t)kS * 4);
  float* g2    = (float*)carve((size_t)kS * 4);
  float* sumg  = (float*)carve((size_t)kE * 4);
  int*   cnt1  = (int*)  carve((size_t)kE * 4);
  int*   slot1 = (int*)  carve((size_t)kS * 4);
  int*   slot2 = (int*)  carve((size_t)kS * 4);
  float* cw1   = (float*)carve((size_t)kS * 4);
  float* cw2   = (float*)carve((size_t)kS * 4);
  int*   src   = (int*)  carve((size_t)kE * kCap * 4);
  unsigned short* xe  = (unsigned short*)carve((size_t)kE * kCap * kM * 2);
  unsigned short* w1t = (unsigned short*)carve((size_t)kE * kM * kH * 2);  // [E][H][M]
  unsigned short* w2t = (unsigned short*)carve((size_t)kE * kH * kM * 2);  // [E][M][H]
  unsigned short* hb  = (unsigned short*)carve((size_t)kE * kCap * kH * 2);
  float* eo = (float*)carve((size_t)kE * kCap * kM * 4);

  init_kernel<<<1, 64, 0, stream>>>(sumg, cnt1);
  // W1 [K=1024, N=4096] -> W1^T [4096, 1024] bf16 per expert
  cvtT_bf16<<<dim3(kH / 64, kM / 64, kE), 256, 0, stream>>>(w1, w1t, kM, kH);
  // W2 [K=4096, N=1024] -> W2^T [1024, 4096] bf16 per expert
  cvtT_bf16<<<dim3(kM / 64, kH / 64, kE), 256, 0, stream>>>(w2, w2t, kH, kM);

  gate_kernel<<<kS / 8, 256, 0, stream>>>(x, wg, idx1, idx2, g1, g2, sumg, cnt1);
  scan_kernel<<<1, 1, 0, stream>>>(idx1, idx2, g1, g2, sumg, cnt1,
                                   slot1, slot2, cw1, cw2, src,
                                   out + (size_t)kS * kM);
  gather_kernel<<<kE * kCap, 256, 0, stream>>>(x, src, xe);

  // GEMM1: h = relu(Xe @ W1 + b1)   [E, 2048, 4096] bf16
  moe_gemm<true><<<dim3(kH / 128, kCap / 128, kE), 256, 0, stream>>>(
      xe, w1t, b1, (void*)hb, kCap, kM, kH);
  // GEMM2: eo = h @ W2 + b2         [E, 2048, 1024] fp32
  moe_gemm<false><<<dim3(kM / 128, kCap / 128, kE), 256, 0, stream>>>(
      hb, w2t, b2, (void*)eo, kCap, kH, kM);

  combine_kernel<<<kS, 256, 0, stream>>>(eo, slot1, slot2, cw1, cw2, out);
}
